// Agent57IndustrialModel_88725434401310
// MI455X (gfx1250) — matrix operand order repaired and checked by reading, not compiled, not run on hardware
//
#include <hip/hip_runtime.h>
#include <hip/hip_bf16.h>

// ---------------------------------------------------------------------------
// Agent57IndustrialModel forward for MI455X (gfx1250, wave32, WMMA)
// B=8, W=64, T=256, D=256, TI=32, WL=16, WP=24, G=64
// ---------------------------------------------------------------------------

typedef __attribute__((ext_vector_type(16))) _Float16 v16h;
typedef __attribute__((ext_vector_type(8)))  _Float16 v8h;
typedef __attribute__((ext_vector_type(8)))  float    v8f;

union AFrag { v16h v; v8h h[2]; };
union CFrag { v8f  v; float f[8]; };

// ---- f32 -> f16 row-major with K zero-padding ------------------------------
__global__ void cvt_pad_kernel(const float* __restrict__ src, _Float16* __restrict__ dst,
                               int M, int K, int Kpad) {
  int i = blockIdx.x * blockDim.x + threadIdx.x;
  if (i >= M * Kpad) return;
  int r = i / Kpad, c = i - r * Kpad;
  dst[i] = (c < K) ? (_Float16)src[r * K + c] : (_Float16)0.f;
}

// ---- weight (K,N) f32 -> B-fragment-ordered f16 tiles ----------------------
// Layout: tile t=(kt*NT+nt): 32 lanes x 16 halves contiguous.
// lane l holds column n = nt*16 + (l&15), K = kt*32 + (l>>4)*16 + h  (h=0..15)
__global__ void prep_b_kernel(const float* __restrict__ w, _Float16* __restrict__ bp,
                              int K, int N, int KT) {
  int NT = N >> 4;
  int i = blockIdx.x * blockDim.x + threadIdx.x;
  int total = KT * NT * 512;
  if (i >= total) return;
  int h    = i & 15;
  int lane = (i >> 4) & 31;
  int tile = i >> 9;
  int nt = tile % NT, kt = tile / NT;
  int n = (nt << 4) + (lane & 15);
  int k = (kt << 5) + ((lane >> 4) << 4) + h;
  bp[i] = (k < K) ? (_Float16)w[k * N + n] : (_Float16)0.f;
}

// ---- WMMA GEMM: out = act( A(MxKpad,f16) @ Bprep(Kpad x N) + bias ) --------
// One wave -> one 16x16 tile. Writes f16 (strided/offset) and/or f32 (dense).
__global__ void gemm16_kernel(const _Float16* __restrict__ A,
                              const _Float16* __restrict__ Bp,
                              const float* __restrict__ bias,
                              _Float16* __restrict__ outH,
                              float* __restrict__ outF,
                              int M, int N, int Kpad, int ldoH, int ocolH, int relu) {
  const int NT = N >> 4;
  const int KT = Kpad >> 5;
  int wave = (blockIdx.x * blockDim.x + threadIdx.x) >> 5;
  int lane = threadIdx.x & 31;
  int tiles = (M >> 4) * NT;
  if (wave >= tiles) return;               // wave-uniform: EXEC stays all-ones
  int mt = wave / NT, nt = wave % NT;
  int m    = (mt << 4) + (lane & 15);
  int koff = (lane >> 4) << 4;             // 0 or 16 (A-frag lane-group K offset)
  const _Float16* arow = A + (size_t)m * Kpad;

  v8f acc = {};
  for (int kt = 0; kt < KT; ++kt) {
    int k0 = kt << 5;
    AFrag a;
    a.h[0] = *(const v8h*)(arow + k0 + koff);        // K = k0+koff .. +7
    a.h[1] = *(const v8h*)(arow + k0 + koff + 8);    // K = k0+koff+8 .. +15
    v16h b = *(const v16h*)(Bp + (((size_t)kt * NT + nt) * 32 + lane) * 16);
    acc = __builtin_amdgcn_wmma_f32_16x16x32_f16(
        false, a.v, false, b, (short)0, acc, false, false);
  }

  int col = (nt << 4) + (lane & 15);
  float bv = bias ? bias[col] : 0.f;
  int rbase = (mt << 4) + ((lane >> 4) << 3);        // C/D: vgpr j -> row j (+8 for hi lanes)
  CFrag c; c.v = acc;
#pragma unroll
  for (int j = 0; j < 8; ++j) {
    float v = c.f[j] + bv;
    if (relu) v = fmaxf(v, 0.f);
    int r = rbase + j;
    if (outF) outF[(size_t)r * N + col] = v;
    if (outH) outH[(size_t)r * ldoH + ocolH + col] = (_Float16)v;
  }
}

// ---- LayerNorm (in-place, D=256, wave per row) + actor-decomposition stats -
__global__ void ln_stats_kernel(float* __restrict__ x,
                                const float* __restrict__ g, const float* __restrict__ bt,
                                const float* __restrict__ fng_seg, const float* __restrict__ aw_seg,
                                float* __restrict__ S1, float* __restrict__ S2,
                                float* __restrict__ C, int rows) {
  int wave = (blockIdx.x * blockDim.x + threadIdx.x) >> 5;
  int lane = threadIdx.x & 31;
  if (wave >= rows) return;
  float* row = x + (size_t)wave * 256;
  float v[8], s = 0.f, s2 = 0.f;
#pragma unroll
  for (int j = 0; j < 8; ++j) {
    float t = row[lane + 32 * j];
    v[j] = t; s += t; s2 += t * t;
  }
  for (int off = 16; off; off >>= 1) {
    s  += __shfl_xor(s,  off, 32);
    s2 += __shfl_xor(s2, off, 32);
  }
  float mu  = s * (1.f / 256.f);
  float var = s2 * (1.f / 256.f) - mu * mu;
  float inv = rsqrtf(var + 1e-5f);
  float zs = 0.f, zs2 = 0.f, zc = 0.f;
#pragma unroll
  for (int j = 0; j < 8; ++j) {
    int d = lane + 32 * j;
    float z = (v[j] - mu) * inv * g[d] + bt[d];
    row[d] = z;
    zs += z; zs2 += z * z; zc += fng_seg[d] * aw_seg[d] * z;
  }
  for (int off = 16; off; off >>= 1) {
    zs  += __shfl_xor(zs,  off, 32);
    zs2 += __shfl_xor(zs2, off, 32);
    zc  += __shfl_xor(zc,  off, 32);
  }
  if (lane == 0) { S1[wave] = zs; S2[wave] = zs2; C[wave] = zc; }
}

// ---- scalar constants: Gsum = sum(fn_g*aw), Cb = sum(fn_b*aw)+actor_b ------
__global__ void scal_kernel(const float* __restrict__ fn_g, const float* __restrict__ fn_b,
                            const float* __restrict__ aw, const float* __restrict__ ab,
                            float* __restrict__ scal) {
  __shared__ float sg[256], sb[256];
  int t = threadIdx.x;
  float g = 0.f, b = 0.f;
  for (int i = t; i < 768; i += 256) { g += fn_g[i] * aw[i]; b += fn_b[i] * aw[i]; }
  sg[t] = g; sb[t] = b; __syncthreads();
  for (int s = 128; s; s >>= 1) {
    if (t < s) { sg[t] += sg[t + s]; sb[t] += sb[t + s]; }
    __syncthreads();
  }
  if (t == 0) { scal[0] = sg[0]; scal[1] = sb[0] + ab[0]; }
}

// ---- global head: g_feat = relu(gc@W1+b1)@W2+b2 ; also its stats -----------
__global__ void gfeat_kernel(const float* __restrict__ gc,
                             const float* __restrict__ w1, const float* __restrict__ b1,
                             const float* __restrict__ w2, const float* __restrict__ b2,
                             const float* __restrict__ fn_g, const float* __restrict__ aw,
                             float* __restrict__ vci, float* __restrict__ Sg) {
  int b = blockIdx.x, d = threadIdx.x;
  __shared__ float h1[256];
  __shared__ float r1[256], r2[256], r3[256];
  float s = b1[d];
  for (int i = 0; i < 64; ++i) s += gc[b * 64 + i] * w1[i * 256 + d];
  h1[d] = fmaxf(s, 0.f);
  __syncthreads();
  float g = b2[d];
  for (int i = 0; i < 256; ++i) g += h1[i] * w2[i * 256 + d];
  vci[b * 768 + 512 + d] = g;
  r1[d] = g; r2[d] = g * g; r3[d] = fn_g[512 + d] * aw[512 + d] * g;
  __syncthreads();
  for (int s2 = 128; s2; s2 >>= 1) {
    if (d < s2) { r1[d] += r1[d + s2]; r2[d] += r2[d + s2]; r3[d] += r3[d + s2]; }
    __syncthreads();
  }
  if (d == 0) { Sg[b] = r1[0]; Sg[8 + b] = r2[0]; Sg[16 + b] = r3[0]; }
}

// ---- masked mean pool over T, mean pool over W -> vci segments 0/1 ---------
__global__ void pool_kernel(const float* __restrict__ t_feat, const float* __restrict__ w_feat,
                            const float* __restrict__ mask, float* __restrict__ vci) {
  int b = blockIdx.x, d = threadIdx.x;
  __shared__ float ms[256];
  ms[d] = mask[b * 256 + d];
  __syncthreads();
  for (int s = 128; s; s >>= 1) {
    if (d < s) ms[d] += ms[d + s];
    __syncthreads();
  }
  float inv = 1.f / fmaxf(ms[0], 1.f);
  float ts = 0.f;
  for (int t = 0; t < 256; ++t)
    ts += mask[b * 256 + t] * t_feat[((size_t)b * 256 + t) * 256 + d];
  vci[b * 768 + d] = ts * inv;
  float wsum = 0.f;
  for (int wi = 0; wi < 64; ++wi)
    wsum += w_feat[((size_t)b * 64 + wi) * 256 + d];
  vci[b * 768 + 256 + d] = wsum * (1.f / 64.f);
}

// ---- actor: decomposed fusion-LN + dot + sigmoid; plus std -----------------
__global__ void actor_kernel(const float* __restrict__ S1t, const float* __restrict__ S2t,
                             const float* __restrict__ Ct,
                             const float* __restrict__ S1w, const float* __restrict__ S2w,
                             const float* __restrict__ Cw,
                             const float* __restrict__ Sg, const float* __restrict__ scal,
                             const float* __restrict__ log_std, float* __restrict__ out) {
  int idx = blockIdx.x * blockDim.x + threadIdx.x;
  if (idx >= 8 * 64 * 256) return;
  int b = idx >> 14;
  int r = idx & 16383;
  int w = r >> 8, t = r & 255;
  float s1 = S1w[b * 64 + w] + S1t[b * 256 + t] + Sg[b];
  float s2 = S2w[b * 64 + w] + S2t[b * 256 + t] + Sg[8 + b];
  float c  = Cw[b * 64 + w]  + Ct[b * 256 + t]  + Sg[16 + b];
  float mu  = s1 * (1.f / 768.f);
  float var = s2 * (1.f / 768.f) - mu * mu;
  float inv = rsqrtf(var + 1e-5f);
  float logit = inv * (c - mu * scal[0]) + scal[1];
  out[idx] = 1.f / (1.f + __expf(-logit));
  float ls = log_std[w * 256 + t];
  ls = fminf(fmaxf(ls, -4.f), 1.f);
  out[131072 + idx] = __expf(ls);
}

// ---- value heads: v = relu(vci@W1+b1)@W2+b2 --------------------------------
__global__ void value_kernel(const float* __restrict__ vci,
                             const float* __restrict__ vu_w1, const float* __restrict__ vu_b1,
                             const float* __restrict__ vu_w2, const float* __restrict__ vu_b2,
                             const float* __restrict__ vc_w1, const float* __restrict__ vc_b1,
                             const float* __restrict__ vc_w2, const float* __restrict__ vc_b2,
                             float* __restrict__ out) {
  int b = blockIdx.x & 7, head = blockIdx.x >> 3, d = threadIdx.x;
  const float* w1 = head ? vc_w1 : vu_w1;
  const float* b1 = head ? vc_b1 : vu_b1;
  const float* w2 = head ? vc_w2 : vu_w2;
  const float* b2 = head ? vc_b2 : vu_b2;
  const float* x = vci + b * 768;
  float s = b1[d];
  for (int i = 0; i < 768; ++i) s += x[i] * w1[i * 256 + d];
  float h = fmaxf(s, 0.f) * w2[d];
  __shared__ float rr[256];
  rr[d] = h; __syncthreads();
  for (int s2 = 128; s2; s2 >>= 1) {
    if (d < s2) rr[d] += rr[d + s2];
    __syncthreads();
  }
  if (d == 0) out[262144 + head * 8 + b] = rr[0] + b2[0];
}

// ---------------------------------------------------------------------------
extern "C" void kernel_launch(void* const* d_in, const int* in_sizes, int n_in,
                              void* d_out, int out_size, void* d_ws, size_t ws_size,
                              hipStream_t stream) {
  (void)in_sizes; (void)n_in; (void)out_size; (void)ws_size;

  const float* task_obs = (const float*)d_in[0];
  const float* wloads   = (const float*)d_in[1];
  const float* wprof    = (const float*)d_in[2];
  const float* gctx     = (const float*)d_in[3];
  const float* vmask    = (const float*)d_in[4];
  const float *te_w1=(const float*)d_in[5],  *te_b1=(const float*)d_in[6];
  const float *te_w2=(const float*)d_in[7],  *te_b2=(const float*)d_in[8];
  const float *wle_w1=(const float*)d_in[9], *wle_b1=(const float*)d_in[10];
  const float *wle_w2=(const float*)d_in[11],*wle_b2=(const float*)d_in[12];
  const float *wpe_w1=(const float*)d_in[13],*wpe_b1=(const float*)d_in[14];
  const float *wpe_w2=(const float*)d_in[15],*wpe_b2=(const float*)d_in[16];
  const float *ft_w=(const float*)d_in[17],  *ft_b=(const float*)d_in[18];
  const float *ft_g=(const float*)d_in[19],  *ft_bt=(const float*)d_in[20];
  const float *fw_w=(const float*)d_in[21],  *fw_b=(const float*)d_in[22];
  const float *fw_g=(const float*)d_in[23],  *fw_bt=(const float*)d_in[24];
  const float *g_w1=(const float*)d_in[25],  *g_b1=(const float*)d_in[26];
  const float *g_w2=(const float*)d_in[27],  *g_b2=(const float*)d_in[28];
  const float *fn_g=(const float*)d_in[29],  *fn_b=(const float*)d_in[30];
  const float *actor_w=(const float*)d_in[31],*actor_b=(const float*)d_in[32];
  const float *log_std=(const float*)d_in[33];
  const float *vu_w1=(const float*)d_in[34], *vu_b1=(const float*)d_in[35];
  const float *vu_w2=(const float*)d_in[36], *vu_b2=(const float*)d_in[37];
  const float *vc_w1=(const float*)d_in[38], *vc_b1=(const float*)d_in[39];
  const float *vc_w2=(const float*)d_in[40], *vc_b2=(const float*)d_in[41];

  float* out = (float*)d_out;

  // -------- workspace bump allocator (256-B aligned) --------
  char* base = (char*)d_ws;
  size_t off = 0;
  auto alloc = [&](size_t bytes) -> void* {
    off = (off + 255) & ~(size_t)255;
    void* p = base + off;
    off += bytes;
    return p;
  };
  _Float16* task16  = (_Float16*)alloc(2048 * 32 * 2);
  _Float16* loads16 = (_Float16*)alloc(512 * 32 * 2);
  _Float16* prof16  = (_Float16*)alloc(512 * 32 * 2);
  _Float16* h1t     = (_Float16*)alloc(2048 * 256 * 2);
  _Float16* h2t     = (_Float16*)alloc(2048 * 256 * 2);
  _Float16* wlh     = (_Float16*)alloc(512 * 256 * 2);
  _Float16* wph     = (_Float16*)alloc(512 * 256 * 2);
  _Float16* cat     = (_Float16*)alloc(512 * 512 * 2);
  _Float16* te_w1p  = (_Float16*)alloc(32 * 256 * 2);
  _Float16* te_w2p  = (_Float16*)alloc(256 * 256 * 2);
  _Float16* ft_wp   = (_Float16*)alloc(256 * 256 * 2);
  _Float16* wle_w1p = (_Float16*)alloc(32 * 256 * 2);
  _Float16* wle_w2p = (_Float16*)alloc(256 * 256 * 2);
  _Float16* wpe_w1p = (_Float16*)alloc(32 * 256 * 2);
  _Float16* wpe_w2p = (_Float16*)alloc(256 * 256 * 2);
  _Float16* fw_wp   = (_Float16*)alloc(512 * 256 * 2);
  float* preT = (float*)alloc(2048 * 256 * 4);   // -> t_feat after in-place LN
  float* preW = (float*)alloc(512 * 256 * 4);    // -> w_feat after in-place LN
  float* S1t = (float*)alloc(2048 * 4);
  float* S2t = (float*)alloc(2048 * 4);
  float* Ct  = (float*)alloc(2048 * 4);
  float* S1w = (float*)alloc(512 * 4);
  float* S2w = (float*)alloc(512 * 4);
  float* Cw  = (float*)alloc(512 * 4);
  float* Sg  = (float*)alloc(3 * 8 * 4);
  float* vci = (float*)alloc(8 * 768 * 4);
  float* scal = (float*)alloc(2 * 4);

  auto cld = [](int n, int b) { return (n + b - 1) / b; };
  auto gblocks = [](int M, int N) { return ((M / 16) * (N / 16) + 7) / 8; };

  // activation f16 staging (zero-padded K)
  cvt_pad_kernel<<<cld(2048 * 32, 256), 256, 0, stream>>>(task_obs, task16, 2048, 32, 32);
  cvt_pad_kernel<<<cld(512 * 32, 256), 256, 0, stream>>>(wloads, loads16, 512, 16, 32);
  cvt_pad_kernel<<<cld(512 * 32, 256), 256, 0, stream>>>(wprof, prof16, 512, 24, 32);

  // weight B-fragment staging
  prep_b_kernel<<<cld(1 * 16 * 512, 256), 256, 0, stream>>>(te_w1, te_w1p, 32, 256, 1);
  prep_b_kernel<<<cld(8 * 16 * 512, 256), 256, 0, stream>>>(te_w2, te_w2p, 256, 256, 8);
  prep_b_kernel<<<cld(8 * 16 * 512, 256), 256, 0, stream>>>(ft_w, ft_wp, 256, 256, 8);
  prep_b_kernel<<<cld(1 * 16 * 512, 256), 256, 0, stream>>>(wle_w1, wle_w1p, 16, 256, 1);
  prep_b_kernel<<<cld(8 * 16 * 512, 256), 256, 0, stream>>>(wle_w2, wle_w2p, 256, 256, 8);
  prep_b_kernel<<<cld(1 * 16 * 512, 256), 256, 0, stream>>>(wpe_w1, wpe_w1p, 24, 256, 1);
  prep_b_kernel<<<cld(8 * 16 * 512, 256), 256, 0, stream>>>(wpe_w2, wpe_w2p, 256, 256, 8);
  prep_b_kernel<<<cld(16 * 16 * 512, 256), 256, 0, stream>>>(fw_w, fw_wp, 512, 256, 16);

  scal_kernel<<<1, 256, 0, stream>>>(fn_g, fn_b, actor_w, actor_b, scal);

  // task path (WMMA): 32->256 relu, 256->256, 256->256 relu -> preT (f32)
  gemm16_kernel<<<gblocks(2048, 256), 256, 0, stream>>>(task16, te_w1p, te_b1, h1t, nullptr,
                                                        2048, 256, 32, 256, 0, 1);
  gemm16_kernel<<<gblocks(2048, 256), 256, 0, stream>>>(h1t, te_w2p, te_b2, h2t, nullptr,
                                                        2048, 256, 256, 256, 0, 0);
  gemm16_kernel<<<gblocks(2048, 256), 256, 0, stream>>>(h2t, ft_wp, ft_b, nullptr, preT,
                                                        2048, 256, 256, 0, 0, 1);
  // worker path (WMMA): two encoders into concat buffer, then fusion GEMM
  gemm16_kernel<<<gblocks(512, 256), 256, 0, stream>>>(loads16, wle_w1p, wle_b1, wlh, nullptr,
                                                       512, 256, 32, 256, 0, 1);
  gemm16_kernel<<<gblocks(512, 256), 256, 0, stream>>>(wlh, wle_w2p, wle_b2, cat, nullptr,
                                                       512, 256, 256, 512, 0, 0);
  gemm16_kernel<<<gblocks(512, 256), 256, 0, stream>>>(prof16, wpe_w1p, wpe_b1, wph, nullptr,
                                                       512, 256, 32, 256, 0, 1);
  gemm16_kernel<<<gblocks(512, 256), 256, 0, stream>>>(wph, wpe_w2p, wpe_b2, cat, nullptr,
                                                       512, 256, 256, 512, 256, 0);
  gemm16_kernel<<<gblocks(512, 256), 256, 0, stream>>>(cat, fw_wp, fw_b, nullptr, preW,
                                                       512, 256, 512, 0, 0, 1);

  // LN in-place + actor-decomposition stats (segments: w=0..255, t=256..511, g=512..767)
  ln_stats_kernel<<<2048 / 8, 256, 0, stream>>>(preT, ft_g, ft_bt, fn_g + 256, actor_w + 256,
                                                S1t, S2t, Ct, 2048);
  ln_stats_kernel<<<512 / 8, 256, 0, stream>>>(preW, fw_g, fw_bt, fn_g, actor_w,
                                               S1w, S2w, Cw, 512);

  gfeat_kernel<<<8, 256, 0, stream>>>(gctx, g_w1, g_b1, g_w2, g_b2, fn_g, actor_w, vci, Sg);
  pool_kernel<<<8, 256, 0, stream>>>(preT, preW, vmask, vci);

  actor_kernel<<<131072 / 256, 256, 0, stream>>>(S1t, S2t, Ct, S1w, S2w, Cw, Sg, scal,
                                                 log_std, out);
  value_kernel<<<16, 256, 0, stream>>>(vci, vu_w1, vu_b1, vu_w2, vu_b2,
                                       vc_w1, vc_b1, vc_w2, vc_b2, out);
}
